// Pipe_25305947308850
// MI455X (gfx1250) — matrix operand order repaired and checked by reading, not compiled, not run on hardware
//
#include <hip/hip_runtime.h>
#include <cstdint>
#include <cstddef>

#define BATCH   256
#define IN_D    512
#define OUT_D   512
#define NWORK   512
#define TOPK    154

// ---- CDNA5 WMMA types ----
typedef __attribute__((ext_vector_type(8)))  float          v8f;
typedef __attribute__((ext_vector_type(16))) __bf16         v16bf;
typedef __attribute__((ext_vector_type(16))) unsigned short v16us;

union ABFrag { uint4 q[2]; v16us us; };

__device__ __forceinline__ unsigned short f32_bf16_rne(float f) {
  unsigned int u = __float_as_uint(f);
  u += 0x7FFFu + ((u >> 16) & 1u);     // round-to-nearest-even
  return (unsigned short)(u >> 16);
}

// -------------------------------------------------------------------------
// Kernel 0: zero the output (harness poisons d_out; we accumulate atomically)
// -------------------------------------------------------------------------
__global__ __launch_bounds__(256) void zero_out_kernel(float4* out) {
  out[blockIdx.x * 256 + threadIdx.x] = make_float4(0.f, 0.f, 0.f, 0.f);
}

// -------------------------------------------------------------------------
// Kernel 1: router logits = x @ gate_w^T + gate_b   (tiny: 67M MAC)
// Block: 16 rows x 64 experts tile, both operands staged in LDS.
// -------------------------------------------------------------------------
#define GW_TILE_R 16
#define GW_TILE_E 64
#define LSTR      516   // +4 float pad to dodge bank conflicts

__global__ __launch_bounds__(256) void router_logits_kernel(
    const float* __restrict__ x, const float* __restrict__ gw,
    const float* __restrict__ gb, float* __restrict__ logits) {
  extern __shared__ char smem_l[];
  float* xt = (float*)smem_l;                 // [16][516]
  float* gt = xt + GW_TILE_R * LSTR;          // [64][516]
  const int r0 = blockIdx.x * GW_TILE_R;
  const int e0 = blockIdx.y * GW_TILE_E;
  const int t  = threadIdx.x;

  for (int i = t; i < GW_TILE_R * IN_D / 4; i += 256) {
    int row = (i * 4) / IN_D, col = (i * 4) % IN_D;
    float4 v = *(const float4*)&x[(size_t)(r0 + row) * IN_D + col];
    float* p = &xt[row * LSTR + col];
    p[0] = v.x; p[1] = v.y; p[2] = v.z; p[3] = v.w;
  }
  for (int i = t; i < GW_TILE_E * IN_D / 4; i += 256) {
    int row = (i * 4) / IN_D, col = (i * 4) % IN_D;
    float4 v = *(const float4*)&gw[(size_t)(e0 + row) * IN_D + col];
    float* p = &gt[row * LSTR + col];
    p[0] = v.x; p[1] = v.y; p[2] = v.z; p[3] = v.w;
  }
  __syncthreads();

  float acc[4] = {0.f, 0.f, 0.f, 0.f};
  int rowi[4], ei[4];
#pragma unroll
  for (int i = 0; i < 4; ++i) { int p = t + 256 * i; rowi[i] = p >> 6; ei[i] = p & 63; }
  for (int d = 0; d < IN_D; ++d) {
#pragma unroll
    for (int i = 0; i < 4; ++i)
      acc[i] += xt[rowi[i] * LSTR + d] * gt[ei[i] * LSTR + d];
  }
#pragma unroll
  for (int i = 0; i < 4; ++i)
    logits[(size_t)(r0 + rowi[i]) * NWORK + e0 + ei[i]] = acc[i] + gb[e0 + ei[i]];
}

// -------------------------------------------------------------------------
// Kernel 2: exact top-K (rank by count-greater, lower-index tie-break, same
// as lax.top_k) + softmax over selected -> dense combine matrix w_full[B,E].
// One block per batch row.
// -------------------------------------------------------------------------
__global__ __launch_bounds__(256) void topk_softmax_kernel(
    const float* __restrict__ logits, float* __restrict__ wfull) {
  __shared__ float l[NWORK];
  __shared__ float red[256];
  const int b = blockIdx.x, t = threadIdx.x;
  l[t]       = logits[(size_t)b * NWORK + t];
  l[t + 256] = logits[(size_t)b * NWORK + t + 256];
  __syncthreads();

  const float v1 = l[t], v2 = l[t + 256];
  int rank1 = 0, rank2 = 0;
  for (int j = 0; j < NWORK; ++j) {
    float lj = l[j];
    rank1 += (lj > v1) || (lj == v1 && j < t);
    rank2 += (lj > v2) || (lj == v2 && j < (t + 256));
  }
  // row max (top-1 is always selected, so row max == max of selected)
  red[t] = fmaxf(v1, v2);
  __syncthreads();
  for (int s = 128; s > 0; s >>= 1) { if (t < s) red[t] = fmaxf(red[t], red[t + s]); __syncthreads(); }
  const float mx = red[0];
  __syncthreads();

  const float e1 = (rank1 < TOPK) ? __expf(v1 - mx) : 0.f;
  const float e2 = (rank2 < TOPK) ? __expf(v2 - mx) : 0.f;
  red[t] = e1 + e2;
  __syncthreads();
  for (int s = 128; s > 0; s >>= 1) { if (t < s) red[t] += red[t + s]; __syncthreads(); }
  const float inv = 1.f / red[0];
  wfull[(size_t)b * NWORK + t]       = e1 * inv;
  wfull[(size_t)b * NWORK + t + 256] = e2 * inv;
}

// -------------------------------------------------------------------------
// Kernel 3: main MoE GEMM with bf16 WMMA, double-buffered expert strips.
// Grid: (OUT_D/16 N-strips) x (NWORK/16 expert chunks). Block = 256 = 8 waves.
// Each block covers the FULL batch M=256 so tiles stream from HBM once
// (x re-reads stay L2-resident: x is only 512KB vs 192MB L2).
// Per expert e: global loads for strip e+1 issue BEFORE the K-loop (latency
// hidden under 32 WMMAs), compute runs from LDS buffer e&1, converted data
// lands in buffer (e+1)&1, one barrier per iteration.
// -------------------------------------------------------------------------
#define XSTR   520   // 512 + 8 bf16 pad (16B) -> conflict-free strided b128
#define TSTR   520
#define ECHUNK 16

__global__ __launch_bounds__(256) void moe_wmma_kernel(
    const float* __restrict__ x, const float* __restrict__ tiles,
    const float* __restrict__ wfull, float* __restrict__ out) {
  extern __shared__ char smem_m[];
  unsigned short* xs = (unsigned short*)smem_m;        // [256][520] bf16  (~260KB)
  unsigned short* ts = xs + BATCH * XSTR;              // 2 x [16][520] bf16 (~32.5KB)
  float*          wl = (float*)(ts + 2 * 16 * TSTR);   // [256][16]  f32   (16KB)

  const int t  = threadIdx.x;
  const int n0 = blockIdx.x * 16;
  const int e0 = blockIdx.y * ECHUNK;

  // Stage x as bf16 in LDS (reused across all experts of this block).
  for (int i = t; i < BATCH * IN_D / 4; i += 256) {
    int row = (i * 4) / IN_D, col = (i * 4) % IN_D;
    float4 v = *(const float4*)&x[(size_t)row * IN_D + col];
    unsigned short* p = &xs[row * XSTR + col];
    p[0] = f32_bf16_rne(v.x); p[1] = f32_bf16_rne(v.y);
    p[2] = f32_bf16_rne(v.z); p[3] = f32_bf16_rne(v.w);
  }
  // Stage router weights for this expert chunk.
  for (int i = t; i < BATCH * ECHUNK; i += 256) {
    int row = i / ECHUNK, el = i % ECHUNK;
    wl[row * ECHUNK + el] = wfull[(size_t)row * NWORK + e0 + el];
  }
  // Prologue: fill strip buffer 0 with expert e0.
  {
    const float* tsrc = tiles + ((size_t)e0 * OUT_D + n0) * IN_D;
#pragma unroll
    for (int j = 0; j < 8; ++j) {
      int i = t + j * 256;
      int row = i >> 7, col = (i & 127) * 4;
      float4 v = *(const float4*)&tsrc[(size_t)row * IN_D + col];
      unsigned short* p = &ts[row * TSTR + col];
      p[0] = f32_bf16_rne(v.x); p[1] = f32_bf16_rne(v.y);
      p[2] = f32_bf16_rne(v.z); p[3] = f32_bf16_rne(v.w);
    }
  }
  __syncthreads();

  const int lane = t & 31, wv = t >> 5;
  const int half = lane >> 4, l16 = lane & 15;
  const int mb0 = wv * 32, mb1 = wv * 32 + 16;   // two 16-row M-subtiles per wave

  v8f acc0 = {0.f,0.f,0.f,0.f,0.f,0.f,0.f,0.f};
  v8f acc1 = {0.f,0.f,0.f,0.f,0.f,0.f,0.f,0.f};

  for (int el = 0; el < ECHUNK; ++el) {
    unsigned short* tcur = ts + (el & 1) * 16 * TSTR;
    unsigned short* tnxt = ts + ((el + 1) & 1) * 16 * TSTR;
    const bool have_next = (el + 1 < ECHUNK);

    // Issue global loads for the NEXT expert strip now; LOADcnt lets them
    // complete under the WMMA K-loop below. Also prefetch expert e+2.
    float4 stage[8];
    if (have_next) {
      const float* tnsrc = tiles + ((size_t)(e0 + el + 1) * OUT_D + n0) * IN_D;
      const bool pf = (e0 + el + 2 < NWORK);
#pragma unroll
      for (int j = 0; j < 8; ++j) {
        int i = t + j * 256;
        int row = i >> 7, col = (i & 127) * 4;
        stage[j] = *(const float4*)&tnsrc[(size_t)row * IN_D + col];
        if (pf)   // global_prefetch_b8: expert e+2, two iterations of depth
          __builtin_prefetch(&tnsrc[(size_t)OUT_D * IN_D + (size_t)row * IN_D + col], 0, 1);
      }
    }

    // ---- GEMM: (256x16) += x(256x512) * strip^T, this wave does 32x16 ----
    v8f c0 = {0.f,0.f,0.f,0.f,0.f,0.f,0.f,0.f};
    v8f c1 = {0.f,0.f,0.f,0.f,0.f,0.f,0.f,0.f};
#pragma unroll 4
    for (int kk = 0; kk < IN_D; kk += 32) {
      // A fragments (16-bit A 16x32: lanes 0-15 K={0..7,16..23}, lanes 16-31 K={8..15,24..31})
      const int ka = kk + half * 8;
      ABFrag a0, a1, bf;
      a0.q[0] = *(const uint4*)&xs[(mb0 + l16) * XSTR + ka];
      a0.q[1] = *(const uint4*)&xs[(mb0 + l16) * XSTR + ka + 16];
      a1.q[0] = *(const uint4*)&xs[(mb1 + l16) * XSTR + ka];
      a1.q[1] = *(const uint4*)&xs[(mb1 + l16) * XSTR + ka + 16];
      // B fragment (same K order as A: lanes 0-15 K={0..7,16..23}, lanes 16-31 K={8..15,24..31})
      const int kb = kk + half * 8;
      const uint4* pb = (const uint4*)&tcur[l16 * TSTR + kb];
      bf.q[0] = pb[0]; bf.q[1] = pb[2];

      v16bf av0 = __builtin_bit_cast(v16bf, a0.us);
      v16bf av1 = __builtin_bit_cast(v16bf, a1.us);
      v16bf bv  = __builtin_bit_cast(v16bf, bf.us);
      c0 = __builtin_amdgcn_wmma_f32_16x16x32_bf16(false, av0, false, bv, (short)0, c0, false, false);
      c1 = __builtin_amdgcn_wmma_f32_16x16x32_bf16(false, av1, false, bv, (short)0, c1, false, false);
    }
    // Scale per-row by router weight, accumulate (C layout: VGPR r -> M=r (+8 hi lanes))
#pragma unroll
    for (int r = 0; r < 8; ++r) {
      acc0[r] += wl[(mb0 + r + half * 8) * ECHUNK + el] * c0[r];
      acc1[r] += wl[(mb1 + r + half * 8) * ECHUNK + el] * c1[r];
    }

    // Convert staged f32 -> bf16 into the other buffer.
    if (have_next) {
#pragma unroll
      for (int j = 0; j < 8; ++j) {
        int i = t + j * 256;
        int row = i >> 7, col = (i & 127) * 4;
        unsigned short* p = &tnxt[row * TSTR + col];
        p[0] = f32_bf16_rne(stage[j].x); p[1] = f32_bf16_rne(stage[j].y);
        p[2] = f32_bf16_rne(stage[j].z); p[3] = f32_bf16_rne(stage[j].w);
      }
    }
    __syncthreads();   // buffer swap
  }

  // Cross-chunk reduction: 32 atomic f32 adds per output element total.
#pragma unroll
  for (int r = 0; r < 8; ++r) {
    unsafeAtomicAdd(&out[(size_t)(mb0 + r + half * 8) * OUT_D + n0 + l16], acc0[r]);
    unsafeAtomicAdd(&out[(size_t)(mb1 + r + half * 8) * OUT_D + n0 + l16], acc1[r]);
  }
}

// -------------------------------------------------------------------------
extern "C" void kernel_launch(void* const* d_in, const int* in_sizes, int n_in,
                              void* d_out, int out_size, void* d_ws, size_t ws_size,
                              hipStream_t stream) {
  const float* x     = (const float*)d_in[0];
  const float* gw    = (const float*)d_in[1];
  const float* gb    = (const float*)d_in[2];
  const float* tiles = (const float*)d_in[3];
  float* out = (float*)d_out;

  float* logits = (float*)d_ws;                         // [256][512]
  float* wfull  = logits + (size_t)BATCH * NWORK;       // [256][512]

  const size_t lshm = (size_t)(GW_TILE_R + GW_TILE_E) * LSTR * sizeof(float);
  const size_t mshm = (size_t)(BATCH * XSTR + 2 * 16 * TSTR) * sizeof(unsigned short)
                    + (size_t)BATCH * ECHUNK * sizeof(float);
  hipFuncSetAttribute((const void*)router_logits_kernel,
                      hipFuncAttributeMaxDynamicSharedMemorySize, (int)lshm);
  hipFuncSetAttribute((const void*)moe_wmma_kernel,
                      hipFuncAttributeMaxDynamicSharedMemorySize, (int)mshm);

  zero_out_kernel<<<BATCH * OUT_D / 4 / 256, 256, 0, stream>>>((float4*)out);
  router_logits_kernel<<<dim3(BATCH / GW_TILE_R, NWORK / GW_TILE_E), 256, lshm, stream>>>(
      x, gw, gb, logits);
  topk_softmax_kernel<<<BATCH, 256, 0, stream>>>(logits, wfull);
  moe_wmma_kernel<<<dim3(OUT_D / 16, NWORK / ECHUNK), 256, mshm, stream>>>(
      x, tiles, wfull, out);
}
